// x13_76501957476997
// MI455X (gfx1250) — matrix operand order repaired and checked by reading, not compiled, not run on hardware
//
#include <hip/hip_runtime.h>
#include <hip/hip_bf16.h>
#include <cstdint>

typedef __attribute__((ext_vector_type(16))) _Float16 v16h;
typedef __attribute__((ext_vector_type(8)))  float    v8f;
typedef __attribute__((ext_vector_type(4)))  unsigned int u32x4;
typedef __attribute__((ext_vector_type(8)))  int      i32x8;
typedef __attribute__((ext_vector_type(4)))  int      i32x4;

#define LN_EPS 1e-5f
#define BN_EPS 1e-5f

#define HAS_TDM __has_builtin(__builtin_amdgcn_tensor_load_to_lds)

// ---------------------------------------------------------------------------
// WMMA helpers (CDNA5 gfx1250, wave32)
// ---------------------------------------------------------------------------
__device__ __forceinline__ v8f wmma_f16(v16h a, v16h b, v8f c) {
  return __builtin_amdgcn_wmma_f32_16x16x32_f16(false, a, false, b, (short)0, c,
                                                false, false);
}

// A-fragment (16x32 MxK): lanes 0-15 -> M=lane, K {0..15};
// lanes 16-31 -> same M, K {16..31}.
__device__ __forceinline__ v16h load_frag_a(const _Float16* __restrict__ lds,
                                            int stride, int row0, int k0,
                                            int lane) {
  int m = lane & 15, h = lane >> 4;
  const _Float16* p = lds + (row0 + m) * stride + k0;
  v16h f;
#pragma unroll
  for (int e = 0; e < 16; e += 2) {
    int k = h * 16 + e;
    f[e]     = p[k];
    f[e + 1] = p[k + 1];
  }
  return f;
}

// B-fragment (32x16 KxN), B staged N-major (Bt[n][k]) in LDS.
__device__ __forceinline__ v16h load_frag_b(const _Float16* __restrict__ lds,
                                            int stride, int n0, int k0,
                                            int lane) {
  int n = lane & 15, h = lane >> 4;
  const _Float16* p = lds + (n0 + n) * stride + k0 + h * 16;
  v16h f;
#pragma unroll
  for (int e = 0; e < 16; ++e) f[e] = p[e];
  return f;
}

// ---------------------------------------------------------------------------
// Tensor Data Mover: 2D f16 tile (tile_k x tile_rows) global -> LDS with
// per-row padding of 4 DWORDs after every 16 DWORDs (row pitch 40 halves).
// ---------------------------------------------------------------------------
#if HAS_TDM
__device__ __forceinline__ void tdm_load_tile_f16(const _Float16* gsrc,
                                                  unsigned lds_off, int tile_k,
                                                  int tile_rows, int stride,
                                                  int td0, int td1) {
  unsigned long long ga = (unsigned long long)(uintptr_t)gsrc;
  u32x4 g0;
  g0[0] = 1u;                                                // count=1
  g0[1] = lds_off;                                           // lds_addr
  g0[2] = (unsigned)(ga & 0xffffffffull);                    // global[31:0]
  g0[3] = (unsigned)((ga >> 32) & 0x1ffffffull) | (2u << 30);  // [56:32]|type=2
  i32x8 g1;
  // data_size=2B | pad_enable | pad_interval=16dw(code3) | pad_amount=4dw(code3)
  g1[0] = (1 << 16) | (1 << 20) | (3 << 22) | (3 << 25);
  g1[1] = (int)((unsigned)(td0 & 0xffff) << 16);             // tensor_dim0 lo
  g1[2] = (int)(((unsigned)(td0 >> 16) & 0xffffu) |
                ((unsigned)(td1 & 0xffff) << 16));           // dim0 hi|dim1 lo
  g1[3] = (int)(((unsigned)(td1 >> 16) & 0xffffu) |
                ((unsigned)(tile_k & 0xffff) << 16));        // dim1 hi|tile0
  g1[4] = (int)((unsigned)(tile_rows & 0xffff));             // tile_dim1
  g1[5] = stride;                                            // dim0_stride lo
  g1[6] = 0;
  g1[7] = 0;
  i32x4 z4 = {0, 0, 0, 0};
  i32x8 z8 = {0, 0, 0, 0, 0, 0, 0, 0};
  __builtin_amdgcn_tensor_load_to_lds(g0, g1, z4, z4, z8, 0);
}
#endif

// ---------------------------------------------------------------------------
// Weight conversion kernels (run once per launch)
// ---------------------------------------------------------------------------
__global__ __launch_bounds__(256) void cvt_f32_f16(const float* __restrict__ s,
                                                   _Float16* __restrict__ d,
                                                   int n) {
  int i = blockIdx.x * 256 + threadIdx.x;
  if (i < n) d[i] = (_Float16)s[i];
}

// [co,ci,3,3] f32 -> [tap][co][ci] f16 (tap-major => contiguous B tiles)
__global__ __launch_bounds__(256) void cvt_conv_w(const float* __restrict__ s,
                                                  _Float16* __restrict__ d) {
  int i = blockIdx.x * 256 + threadIdx.x;  // over 256*256*9
  int tap = i % 9, cc = i / 9;
  d[(size_t)tap * 65536 + cc] = (_Float16)s[i];
}

// ---------------------------------------------------------------------------
// LayerNorm over C=256, one row per block, f16 out
// ---------------------------------------------------------------------------
__global__ __launch_bounds__(256) void layernorm_to_f16(
    const float* __restrict__ x, const float* __restrict__ g,
    const float* __restrict__ b, _Float16* __restrict__ out) {
  int row = blockIdx.x, tid = threadIdx.x;
  size_t idx = (size_t)row * 256 + tid;
  float v = x[idx];
  float s = v, q = v * v;
#pragma unroll
  for (int o = 16; o > 0; o >>= 1) {
    s += __shfl_xor(s, o);
    q += __shfl_xor(q, o);
  }
  __shared__ float ws[8], wq[8];
  int lane = tid & 31, wave = tid >> 5;
  if (lane == 0) { ws[wave] = s; wq[wave] = q; }
  __syncthreads();
  if (tid == 0) {
    float S = 0.f, Q = 0.f;
#pragma unroll
    for (int i = 0; i < 8; ++i) { S += ws[i]; Q += wq[i]; }
    ws[0] = S; wq[0] = Q;
  }
  __syncthreads();
  float mean = ws[0] * (1.f / 256.f);
  float var  = wq[0] * (1.f / 256.f) - mean * mean;
  float r = rsqrtf(var + LN_EPS);
  out[idx] = (_Float16)((v - mean) * r * g[tid] + b[tid]);
}

// ---------------------------------------------------------------------------
// Generic WMMA GEMM, 128x64 tile, TDM double-buffered tile loads.
//   Each of 8 waves: one 16-row M strip x four 16-col N strips (4 WMMA/step).
//   A [M,K] f16, W [N,K] f16. epi: 0=f16 ; 1=f16 gelu(acc+bias) ;
//   2=f32 acc+bias+resid.  Dynamic LDS: 2*(128*40 + 64*40) halves = 30 KB.
// ---------------------------------------------------------------------------
#define GMT 128
#define GNT 64
#define GKT 32
#define SLDA 40
#define SLDB 40
#define GEMM_LDS_BYTES ((GMT * SLDA + GNT * SLDB) * 2 * 2)

__global__ __launch_bounds__(256) void gemm_wmma(
    const _Float16* __restrict__ A, const _Float16* __restrict__ W,
    const float* __restrict__ bias, const float* __restrict__ resid,
    float* __restrict__ o32, _Float16* __restrict__ o16, int M, int N, int K,
    int epi) {
  extern __shared__ _Float16 smem[];
  const int ASZ = GMT * SLDA, BSZ = GNT * SLDB;
  const int BUFSZ = ASZ + BSZ;  // halves per double-buffer slot
  int tid = threadIdx.x, lane = tid & 31, wave = tid >> 5;
  int m0 = blockIdx.x * GMT, n0 = blockIdx.y * GNT;
  const _Float16* abase = A + (size_t)m0 * K;
  const _Float16* wbase = W + (size_t)n0 * K;
  v8f acc[4] = {v8f{}, v8f{}, v8f{}, v8f{}};
  int nk = K / GKT;
#if HAS_TDM
  if (wave == 0) {
    tdm_load_tile_f16(abase, 0u, GKT, GMT, K, K, M);
    tdm_load_tile_f16(wbase, (unsigned)(ASZ * 2), GKT, GNT, K, K, N);
  }
#endif
  for (int ki = 0; ki < nk; ++ki) {
    int buf = ki & 1;
    _Float16* Ab = smem + buf * BUFSZ;
    _Float16* Bb = Ab + ASZ;
#if HAS_TDM
    if (wave == 0) {
      if (ki + 1 < nk) {
        unsigned nb = (unsigned)((buf ^ 1) * BUFSZ * 2);
        tdm_load_tile_f16(abase + (ki + 1) * GKT, nb, GKT, GMT, K, K, M);
        tdm_load_tile_f16(wbase + (ki + 1) * GKT, nb + (unsigned)(ASZ * 2), GKT,
                          GNT, K, K, N);
        __builtin_amdgcn_s_wait_tensorcnt(2);  // current buf complete
      } else {
        __builtin_amdgcn_s_wait_tensorcnt(0);
      }
    }
    __syncthreads();
#else
    {  // VALU fallback staging
      int ar = tid >> 1, asg = (tid & 1) * 16;
      const _Float16* as = abase + (size_t)ar * K + ki * GKT + asg;
      _Float16* ad = Ab + ar * SLDA + asg;
#pragma unroll
      for (int j = 0; j < 16; ++j) ad[j] = as[j];
      int br = tid >> 2, bsg = (tid & 3) * 8;
      const _Float16* bs = wbase + (size_t)br * K + ki * GKT + bsg;
      _Float16* bd = Bb + br * SLDB + bsg;
#pragma unroll
      for (int j = 0; j < 8; ++j) bd[j] = bs[j];
    }
    __syncthreads();
#endif
    v16h af = load_frag_a(Ab, SLDA, wave * 16, 0, lane);
#pragma unroll
    for (int t = 0; t < 4; ++t) {
      v16h bf = load_frag_b(Bb, SLDB, t * 16, 0, lane);
      acc[t] = wmma_f16(af, bf, acc[t]);
    }
    __syncthreads();
  }
  int cn = lane & 15, ch = lane >> 4;
#pragma unroll
  for (int t = 0; t < 4; ++t) {
    int gcol = n0 + t * 16 + cn;
#pragma unroll
    for (int r = 0; r < 8; ++r) {
      int grow = m0 + wave * 16 + r + 8 * ch;
      size_t idx = (size_t)grow * N + gcol;
      float v = acc[t][r];
      if (epi == 1) {
        v += bias[gcol];
        v = 0.5f * v * (1.f + erff(v * 0.70710678118654752f));  // exact GELU
        o16[idx] = (_Float16)v;
      } else if (epi == 2) {
        o32[idx] = v + bias[gcol] + resid[idx];
      } else {
        o16[idx] = (_Float16)v;
      }
    }
  }
}

// ---------------------------------------------------------------------------
// 3x3 SAME conv (48x48, C=256->256) + eval-BN as 9 shifted GEMM taps.
// Weights pre-transformed to [tap][co][ci] f16 (contiguous B tiles).
// ---------------------------------------------------------------------------
#define CMT 64
__global__ __launch_bounds__(256) void conv3x3_bn_wmma(
    const _Float16* __restrict__ xn, const _Float16* __restrict__ Wt,
    const float* __restrict__ bng, const float* __restrict__ bnb,
    _Float16* __restrict__ out) {
  __shared__ _Float16 As[CMT * SLDA];
  __shared__ _Float16 Bs[GNT * SLDB];
  int tid = threadIdx.x, lane = tid & 31, wave = tid >> 5;
  int m0 = blockIdx.x * CMT;  // token within batch
  int n0 = blockIdx.y * GNT;  // out channel
  int b  = blockIdx.z;
  int msub = wave & 3, nsub = (wave >> 2) * 2;
  v8f acc0 = {}, acc1 = {};
  int lrow = tid >> 2, lseg = (tid & 3) * 8;
  int py = (m0 + lrow) / 48, px = (m0 + lrow) % 48;
  for (int tap = 0; tap < 9; ++tap) {
    int dy = tap / 3 - 1, dx = tap % 3 - 1;
    int sy = py + dy, sx = px + dx;
    bool ok = ((unsigned)sy < 48u) && ((unsigned)sx < 48u);
    const _Float16* asrc = xn + ((size_t)b * 2304 + sy * 48 + sx) * 256 + lseg;
    const _Float16* wsrc =
        Wt + (size_t)tap * 65536 + (size_t)(n0 + lrow) * 256 + lseg;
    for (int k0 = 0; k0 < 256; k0 += GKT) {
      _Float16* ad = As + lrow * SLDA + lseg;
      if (ok) {
#pragma unroll
        for (int j = 0; j < 8; ++j) ad[j] = asrc[k0 + j];
      } else {
#pragma unroll
        for (int j = 0; j < 8; ++j) ad[j] = (_Float16)0.f;
      }
      _Float16* bd = Bs + lrow * SLDB + lseg;
#pragma unroll
      for (int j = 0; j < 8; ++j) bd[j] = wsrc[k0 + j];
      if (ok && k0 + GKT < 256) __builtin_prefetch(asrc + k0 + GKT, 0, 0);
      __syncthreads();
      v16h af = load_frag_a(As, SLDA, msub * 16, 0, lane);
      v16h b0 = load_frag_b(Bs, SLDB, nsub * 16, 0, lane);
      v16h b1 = load_frag_b(Bs, SLDB, (nsub + 1) * 16, 0, lane);
      acc0 = wmma_f16(af, b0, acc0);
      acc1 = wmma_f16(af, b1, acc1);
      __syncthreads();
    }
  }
  float bnc = rsqrtf(1.f + BN_EPS);
  int cn = lane & 15, ch = lane >> 4;
#pragma unroll
  for (int p = 0; p < 2; ++p) {
    v8f acc = p ? acc1 : acc0;
    int gcol = n0 + (nsub + p) * 16 + cn;
    float sc = bng[gcol] * bnc, sh = bnb[gcol];
#pragma unroll
    for (int r = 0; r < 8; ++r) {
      int grow = m0 + msub * 16 + r + 8 * ch;
      out[((size_t)b * 2304 + grow) * 256 + gcol] = (_Float16)(acc[r] * sc + sh);
    }
  }
}

// ---------------------------------------------------------------------------
// Flash attention: block = (batch, head, 64-query tile); 36 kv tiles of 64.
// ---------------------------------------------------------------------------
#define ALDQ 40
#define ALDS 72

__global__ __launch_bounds__(256) void attn_wmma(
    const _Float16* __restrict__ qh, const _Float16* __restrict__ kh,
    const _Float16* __restrict__ vh, _Float16* __restrict__ o) {
  __shared__ _Float16 Qs[64 * ALDQ];
  __shared__ _Float16 Ks[64 * ALDQ];
  __shared__ _Float16 Vt[32 * ALDS];  // [d][key]
  __shared__ float    Ss[64 * ALDS];
  __shared__ _Float16 Ps[64 * ALDS];
  __shared__ float mrow[64], lrow_s[64], arow[64];
  int tid = threadIdx.x, lane = tid & 31, wave = tid >> 5;
  int q0 = blockIdx.x * 64, head = blockIdx.y, b = blockIdx.z;
  size_t base = ((size_t)b * 2304) * 256 + (size_t)head * 32;
  int lr = tid >> 2, lseg = (tid & 3) * 8;
  {
    const _Float16* src = qh + base + (size_t)(q0 + lr) * 256 + lseg;
    _Float16* dst = Qs + lr * ALDQ + lseg;
#pragma unroll
    for (int j = 0; j < 8; ++j) dst[j] = src[j];
  }
  if (tid < 64) { mrow[tid] = -3.0e38f; lrow_s[tid] = 0.f; }
  int msub = wave & 3, nsub = (wave >> 2) * 2, onsub = wave >> 2;
  int cn = lane & 15, ch = lane >> 4;
  v8f oacc = {};
  __syncthreads();
  const float scale = 0.0625f;  // 256^-0.5
  for (int kb = 0; kb < 36; ++kb) {
    int kt0 = kb * 64;
    {
      const _Float16* ksrc = kh + base + (size_t)(kt0 + lr) * 256 + lseg;
      _Float16* kd = Ks + lr * ALDQ + lseg;
#pragma unroll
      for (int j = 0; j < 8; ++j) kd[j] = ksrc[j];
      const _Float16* vsrc = vh + base + (size_t)(kt0 + lr) * 256 + lseg;
#pragma unroll
      for (int j = 0; j < 8; ++j) Vt[(lseg + j) * ALDS + lr] = vsrc[j];
    }
    __syncthreads();
    v16h aq = load_frag_a(Qs, ALDQ, msub * 16, 0, lane);
    v16h k0f = load_frag_b(Ks, ALDQ, nsub * 16, 0, lane);
    v16h k1f = load_frag_b(Ks, ALDQ, (nsub + 1) * 16, 0, lane);
    v8f s0 = {}, s1 = {};
    s0 = wmma_f16(aq, k0f, s0);
    s1 = wmma_f16(aq, k1f, s1);
#pragma unroll
    for (int r = 0; r < 8; ++r) {
      int rr = msub * 16 + r + 8 * ch;
      Ss[rr * ALDS + nsub * 16 + cn]       = s0[r] * scale;
      Ss[rr * ALDS + (nsub + 1) * 16 + cn] = s1[r] * scale;
    }
    __syncthreads();
    if (tid < 64) {  // online softmax, one thread per query row
      float* srow = Ss + tid * ALDS;
      float mold = mrow[tid];
      float bm = srow[0];
#pragma unroll 4
      for (int j = 1; j < 64; ++j) bm = fmaxf(bm, srow[j]);
      float mnew = fmaxf(mold, bm);
      float al = __expf(mold - mnew);
      float sum = 0.f;
      _Float16* prow = Ps + tid * ALDS;
#pragma unroll 4
      for (int j = 0; j < 64; ++j) {
        float e = __expf(srow[j] - mnew);
        prow[j] = (_Float16)e;
        sum += e;
      }
      arow[tid] = al;
      lrow_s[tid] = lrow_s[tid] * al + sum;
      mrow[tid] = mnew;
    }
    __syncthreads();
#pragma unroll
    for (int r = 0; r < 8; ++r) oacc[r] *= arow[msub * 16 + r + 8 * ch];
    v16h p0 = load_frag_a(Ps, ALDS, msub * 16, 0, lane);
    v16h v0 = load_frag_b(Vt, ALDS, onsub * 16, 0, lane);
    oacc = wmma_f16(p0, v0, oacc);
    v16h p1 = load_frag_a(Ps, ALDS, msub * 16, 32, lane);
    v16h v1 = load_frag_b(Vt, ALDS, onsub * 16, 32, lane);
    oacc = wmma_f16(p1, v1, oacc);
    __syncthreads();
  }
#pragma unroll
  for (int r = 0; r < 8; ++r) {
    int rr = msub * 16 + r + 8 * ch;
    float inv = 1.f / lrow_s[rr];
    o[base + (size_t)(q0 + rr) * 256 + onsub * 16 + cn] =
        (_Float16)(oacc[r] * inv);
  }
}

// ---------------------------------------------------------------------------
// Host launcher
// ---------------------------------------------------------------------------
extern "C" void kernel_launch(void* const* d_in, const int* in_sizes, int n_in,
                              void* d_out, int out_size, void* d_ws,
                              size_t ws_size, hipStream_t stream) {
  (void)in_sizes; (void)n_in; (void)out_size; (void)ws_size;
  const float* features = (const float*)d_in[0];
  const float* conv_q_w = (const float*)d_in[3];
  const float* bn_q_g   = (const float*)d_in[4];
  const float* bn_q_b   = (const float*)d_in[5];
  const float* conv_k_w = (const float*)d_in[6];
  const float* bn_k_g   = (const float*)d_in[7];
  const float* bn_k_b   = (const float*)d_in[8];
  const float* conv_v_w = (const float*)d_in[9];
  const float* bn_v_g   = (const float*)d_in[10];
  const float* bn_v_b   = (const float*)d_in[11];
  const float* proj_q_w = (const float*)d_in[12];
  const float* proj_k_w = (const float*)d_in[13];
  const float* proj_v_w = (const float*)d_in[14];
  const float* proj_o_w = (const float*)d_in[15];
  const float* proj_o_b = (const float*)d_in[16];
  const float* ln1_g    = (const float*)d_in[17];
  const float* ln1_b    = (const float*)d_in[18];
  const float* ln3_g    = (const float*)d_in[19];
  const float* ln3_b    = (const float*)d_in[20];
  const float* fc1_w    = (const float*)d_in[21];
  const float* fc1_b    = (const float*)d_in[22];
  const float* fc2_w    = (const float*)d_in[23];
  const float* fc2_b    = (const float*)d_in[24];
  float* out = (float*)d_out;

  const size_t TOK = 4ull * 2304 * 256;  // B*T*C
  char* p = (char*)d_ws;
  auto alloc = [&](size_t bytes) {
    void* r = (void*)p;
    p += (bytes + 255) & ~(size_t)255;
    return r;
  };
  _Float16* xn   = (_Float16*)alloc(TOK * 2);
  _Float16* qc   = (_Float16*)alloc(TOK * 2);
  _Float16* kc   = (_Float16*)alloc(TOK * 2);
  _Float16* vc   = (_Float16*)alloc(TOK * 2);
  _Float16* qhB  = (_Float16*)alloc(TOK * 2);
  _Float16* khB  = (_Float16*)alloc(TOK * 2);
  _Float16* vhB  = (_Float16*)alloc(TOK * 2);
  _Float16* att  = (_Float16*)alloc(TOK * 2);
  float*    x1   = (float*)alloc(TOK * 4);
  _Float16* ln3h = (_Float16*)alloc(TOK * 2);
  _Float16* h1   = (_Float16*)alloc(4ull * 2304 * 1024 * 2);
  // f16 weights
  _Float16* wq   = (_Float16*)alloc(9ull * 65536 * 2);
  _Float16* wk   = (_Float16*)alloc(9ull * 65536 * 2);
  _Float16* wv   = (_Float16*)alloc(9ull * 65536 * 2);
  _Float16* pq   = (_Float16*)alloc(65536 * 2);
  _Float16* pk   = (_Float16*)alloc(65536 * 2);
  _Float16* pv   = (_Float16*)alloc(65536 * 2);
  _Float16* po   = (_Float16*)alloc(65536 * 2);
  _Float16* f1w  = (_Float16*)alloc(262144 * 2);
  _Float16* f2w  = (_Float16*)alloc(262144 * 2);

  // 0) weight conversions (f32 -> f16; conv weights tap-major)
  cvt_conv_w<<<2304, 256, 0, stream>>>(conv_q_w, wq);
  cvt_conv_w<<<2304, 256, 0, stream>>>(conv_k_w, wk);
  cvt_conv_w<<<2304, 256, 0, stream>>>(conv_v_w, wv);
  cvt_f32_f16<<<256, 256, 0, stream>>>(proj_q_w, pq, 65536);
  cvt_f32_f16<<<256, 256, 0, stream>>>(proj_k_w, pk, 65536);
  cvt_f32_f16<<<256, 256, 0, stream>>>(proj_v_w, pv, 65536);
  cvt_f32_f16<<<256, 256, 0, stream>>>(proj_o_w, po, 65536);
  cvt_f32_f16<<<1024, 256, 0, stream>>>(fc1_w, f1w, 262144);
  cvt_f32_f16<<<1024, 256, 0, stream>>>(fc2_w, f2w, 262144);

  // 1) LN1 -> f16 tokens
  layernorm_to_f16<<<9216, 256, 0, stream>>>(features, ln1_g, ln1_b, xn);
  // 2) conv+BN for Q,K,V
  dim3 cgrid(36, 4, 4);
  conv3x3_bn_wmma<<<cgrid, 256, 0, stream>>>(xn, wq, bn_q_g, bn_q_b, qc);
  conv3x3_bn_wmma<<<cgrid, 256, 0, stream>>>(xn, wk, bn_k_g, bn_k_b, kc);
  conv3x3_bn_wmma<<<cgrid, 256, 0, stream>>>(xn, wv, bn_v_g, bn_v_b, vc);
  // 3) head projections (TDM GEMM)
  dim3 g256(72, 4);
  gemm_wmma<<<g256, 256, GEMM_LDS_BYTES, stream>>>(
      qc, pq, nullptr, nullptr, nullptr, qhB, 9216, 256, 256, 0);
  gemm_wmma<<<g256, 256, GEMM_LDS_BYTES, stream>>>(
      kc, pk, nullptr, nullptr, nullptr, khB, 9216, 256, 256, 0);
  gemm_wmma<<<g256, 256, GEMM_LDS_BYTES, stream>>>(
      vc, pv, nullptr, nullptr, nullptr, vhB, 9216, 256, 256, 0);
  // 4) flash attention
  attn_wmma<<<dim3(36, 8, 4), 256, 0, stream>>>(qhB, khB, vhB, att);
  // 5) output projection + bias + residual -> x1 (f32)
  gemm_wmma<<<g256, 256, GEMM_LDS_BYTES, stream>>>(
      att, po, proj_o_b, features, x1, nullptr, 9216, 256, 256, 2);
  // 6) LN3 -> f16
  layernorm_to_f16<<<9216, 256, 0, stream>>>(x1, ln3_g, ln3_b, ln3h);
  // 7) MLP
  gemm_wmma<<<dim3(72, 16), 256, GEMM_LDS_BYTES, stream>>>(
      ln3h, f1w, fc1_b, nullptr, nullptr, h1, 9216, 1024, 256, 1);
  gemm_wmma<<<g256, 256, GEMM_LDS_BYTES, stream>>>(h1, f2w, fc2_b, x1, out,
                                                   nullptr, 9216, 256, 1024, 2);
}
